// EdgePredictor_43276090474511
// MI455X (gfx1250) — matrix operand order repaired and checked by reading, not compiled, not run on hardware
//
#include <hip/hip_runtime.h>
#include <hip/hip_bf16.h>

typedef __attribute__((ext_vector_type(16))) _Float16 v16h;
typedef __attribute__((ext_vector_type(8)))  _Float16 v8h;
typedef __attribute__((ext_vector_type(8)))  float    v8f;

#define NPTS 1024
#define DIMD 64
#define ATTNH 256

// ---------------------------------------------------------------------------
// helpers
// ---------------------------------------------------------------------------
__device__ __forceinline__ v16h make16(const _Float16* p0, const _Float16* p1) {
    v8h a = *(const v8h*)p0;
    v8h b = *(const v8h*)p1;
    v16h r;
#pragma unroll
    for (int i = 0; i < 8; ++i) { r[i] = a[i]; r[i + 8] = b[i]; }
    return r;
}

// B tile (K=32 x N=16) from packed buffer: lane l reads 16 contiguous halves.
__device__ __forceinline__ v16h load_btile(const _Float16* base, int t, int lane) {
    const _Float16* p = base + ((size_t)(t * 32 + lane)) * 16;
    return make16(p, p + 8);
}

// A tile (M=16 x K=32) from row-major f16 [16][ldk]; rowbase = ptr to this lane's row.
// lane<16: K = kt*32 + {0..15};  lane>=16: K = kt*32 + {16..31}
__device__ __forceinline__ v16h load_atile(const _Float16* rowbase, int kt, int hi) {
    const _Float16* p = rowbase + kt * 32 + hi * 16;
    return make16(p, p + 8);
}

__device__ __forceinline__ v8f wmma16(v16h a, v16h b, v8f c) {
    return __builtin_amdgcn_wmma_f32_16x16x32_f16(false, a, false, b, (short)0, c,
                                                  false, false);
}

// ---------------------------------------------------------------------------
// weight repack:  f32 W[K x N] row-major  ->  f16 B-lane-packed tiles
// tile t = kt*(N/16)+nt ; within tile: (lane, idx) -> element
// ---------------------------------------------------------------------------
__global__ void pack_b_f16(const float* __restrict__ W, _Float16* __restrict__ out,
                           int K, int Nn) {
    int gid = blockIdx.x * blockDim.x + threadIdx.x;
    int total = (K / 32) * (Nn / 16) * 32 * 16;
    if (gid >= total) return;
    int idx = gid & 15;
    int l   = (gid >> 4) & 31;
    int t   = gid >> 9;
    int NT = Nn / 16;
    int kt = t / NT, nt = t % NT;
    int n = nt * 16 + (l & 15);
    int k = kt * 32 + ((l >> 4) * 16) + idx;
    out[gid] = (_Float16)W[k * Nn + n];
}

// ---------------------------------------------------------------------------
// feats0 = x @ in_w + in_b
// ---------------------------------------------------------------------------
__global__ void inproj_kernel(const float* __restrict__ x, const float* __restrict__ w,
                              const float* __restrict__ b, float* __restrict__ feats) {
    int idx = blockIdx.x * blockDim.x + threadIdx.x;   // 1024*64
    int i = idx >> 6, d = idx & 63;
    feats[idx] = x[2 * i] * w[d] + x[2 * i + 1] * w[64 + d] + b[d];
}

// ---------------------------------------------------------------------------
// qkv = feats @ qkv_w[l]   (split into q,k,v [1024,64] f32)
// ---------------------------------------------------------------------------
__global__ void qkv_kernel(const float* __restrict__ feats, const float* __restrict__ W,
                           float* __restrict__ q, float* __restrict__ k,
                           float* __restrict__ v) {
    int idx = blockIdx.x * blockDim.x + threadIdx.x;   // 1024*192
    int i = idx / 192, m = idx % 192;
    float acc = 0.f;
#pragma unroll 8
    for (int kk = 0; kk < 64; ++kk) acc += feats[i * 64 + kk] * W[kk * 192 + m];
    if (m < 64)       q[i * 64 + m] = acc;
    else if (m < 128) k[i * 64 + (m - 64)] = acc;
    else              v[i * 64 + (m - 128)] = acc;
}

// ---------------------------------------------------------------------------
// fused PointTransformer layer: one block per query i, 4 waves, flash-style
// online softmax over j (per channel d).
// ---------------------------------------------------------------------------
constexpr int SMEM_HALVES = 64 * 64 + 64 * 256 + 256 * 64   // packed pw2/aw1/aw2
                          + 4 * 16 * 64 + 4 * 16 * 256;     // hin / h1 staging
constexpr int SMEM_FLOATS = 192 + 64 + 64 + 256 + 64        // pw1,pb1,pb2,ab1,ab2
                          + 4 * 64 * 3;                     // reduction
constexpr size_t SMEM_BYTES = (size_t)SMEM_HALVES * 2 + (size_t)SMEM_FLOATS * 4;

__global__ void __launch_bounds__(128)
pt_layer_kernel(const float* __restrict__ x,
                const float* __restrict__ q, const float* __restrict__ k,
                const float* __restrict__ v,
                const float* __restrict__ pw1, const float* __restrict__ pb1,
                const float* __restrict__ pb2, const float* __restrict__ ab1,
                const float* __restrict__ ab2,
                const _Float16* __restrict__ packL,   // pw2p | aw1p | aw2p contiguous
                float* __restrict__ feats_out) {
    extern __shared__ char smem[];
    _Float16* s_pw2 = (_Float16*)smem;            // 4096 halves
    _Float16* s_aw1 = s_pw2 + 64 * 64;            // 16384
    _Float16* s_aw2 = s_aw1 + 64 * 256;           // 16384
    _Float16* s_hin = s_aw2 + 256 * 64;           // 4 * 16*64
    _Float16* s_h1  = s_hin + 4 * 16 * 64;        // 4 * 16*256
    float* s_pw1 = (float*)(s_h1 + 4 * 16 * 256); // 192
    float* s_pb1 = s_pw1 + 192;
    float* s_pb2 = s_pb1 + 64;
    float* s_ab1 = s_pb2 + 64;
    float* s_ab2 = s_ab1 + 256;
    float* s_red = s_ab2 + 64;                    // 4*64*3

    const int tid = threadIdx.x;
    // cooperative copy of packed weights (73728 B) to LDS
    {
        const uint4* src = (const uint4*)packL;
        uint4* dst = (uint4*)s_pw2;
        for (int t = tid; t < 4608; t += 128) dst[t] = src[t];
        for (int t = tid; t < 192; t += 128) s_pw1[t] = pw1[t];
        for (int t = tid; t < 256; t += 128) s_ab1[t] = ab1[t];
        if (tid < 64) {
            s_pb1[tid] = pb1[tid];
            s_pb2[tid] = pb2[tid];
            s_ab2[tid] = ab2[tid];
        }
    }
    __syncthreads();

    const int i    = blockIdx.x;
    const int wave = tid >> 5;
    const int lane = tid & 31;
    const int lr   = lane & 15;
    const int hi   = lane >> 4;

    const float xi0 = x[2 * i], xi1 = x[2 * i + 1];
    float qd[4];
#pragma unroll
    for (int nt = 0; nt < 4; ++nt) qd[nt] = q[i * 64 + nt * 16 + lr];

    float m_s[4], s_s[4], a_s[4];
#pragma unroll
    for (int nt = 0; nt < 4; ++nt) { m_s[nt] = -1e30f; s_s[nt] = 0.f; a_s[nt] = 0.f; }

    _Float16* my_hin = s_hin + wave * 16 * 64;
    _Float16* my_h1  = s_h1 + wave * 16 * 256;

    for (int jt = wave; jt < 64; jt += 4) {
        const int jbase = jt * 16;
        const int jrow  = jbase + lr;
        const float rx = xi0 - x[2 * jrow];
        const float ry = xi1 - x[2 * jrow + 1];

        // pos-MLP layer1 (K=2 VALU) directly into A-layout f16
        v16h hA[2];
#pragma unroll
        for (int kt = 0; kt < 2; ++kt) {
#pragma unroll
            for (int e = 0; e < 16; ++e) {
                int kl = hi * 16 + e;
                int p  = kt * 32 + kl;
                float hv = fmaf(rx, s_pw1[p], fmaf(ry, s_pw1[64 + p], s_pb1[p]));
                hA[kt][e] = (_Float16)fmaxf(hv, 0.f);
            }
        }

        // rel_emb = relu(..)@pw2 + pb2  (C layout, kept for vv)
        v8f relC[4];
#pragma unroll
        for (int nt = 0; nt < 4; ++nt) {
            float b = s_pb2[nt * 16 + lr];
            v8f c;
#pragma unroll
            for (int r = 0; r < 8; ++r) c[r] = b;
#pragma unroll
            for (int kt = 0; kt < 2; ++kt)
                c = wmma16(hA[kt], load_btile(s_pw2, kt * 4 + nt, lane), c);
            relC[nt] = c;
        }

        // hin = (q_i - k_j) + rel_emb  -> LDS row-major f16 [16][64]
#pragma unroll
        for (int nt = 0; nt < 4; ++nt) {
#pragma unroll
            for (int r = 0; r < 8; ++r) {
                int row = r + 8 * hi;
                float kv = k[(size_t)(jbase + row) * 64 + nt * 16 + lr];
                float hv = qd[nt] - kv + relC[nt][r];
                my_hin[row * 64 + nt * 16 + lr] = (_Float16)hv;
            }
        }
        v16h hinA[2];
#pragma unroll
        for (int kt = 0; kt < 2; ++kt) hinA[kt] = load_atile(my_hin + lr * 64, kt, hi);

        // attn MLP layer1 (64 -> 256), two phases of 8 col-tiles
#pragma unroll
        for (int ph = 0; ph < 2; ++ph) {
#pragma unroll
            for (int nt8 = 0; nt8 < 8; ++nt8) {
                int nt = ph * 8 + nt8;
                float b = s_ab1[nt * 16 + lr];
                v8f c;
#pragma unroll
                for (int r = 0; r < 8; ++r) c[r] = b;
#pragma unroll
                for (int kt = 0; kt < 2; ++kt)
                    c = wmma16(hinA[kt], load_btile(s_aw1, kt * 16 + nt, lane), c);
#pragma unroll
                for (int r = 0; r < 8; ++r)
                    my_h1[(r + 8 * hi) * 256 + nt * 16 + lr] =
                        (_Float16)fmaxf(c[r], 0.f);
            }
        }

        v16h h1A[8];
#pragma unroll
        for (int kt = 0; kt < 8; ++kt) h1A[kt] = load_atile(my_h1 + lr * 256, kt, hi);

        // attn MLP layer2 (256 -> 64) -> sim, then online softmax update
#pragma unroll
        for (int nt = 0; nt < 4; ++nt) {
            float b = s_ab2[nt * 16 + lr];
            v8f c;
#pragma unroll
            for (int r = 0; r < 8; ++r) c[r] = b;
#pragma unroll
            for (int kt = 0; kt < 8; ++kt)
                c = wmma16(h1A[kt], load_btile(s_aw2, kt * 4 + nt, lane), c);

            float tmax = c[0];
#pragma unroll
            for (int r = 1; r < 8; ++r) tmax = fmaxf(tmax, c[r]);
            tmax = fmaxf(tmax, __shfl_xor(tmax, 16, 32));
            float mnew  = fmaxf(m_s[nt], tmax);
            float alpha = __expf(m_s[nt] - mnew);
            float ps = 0.f, pa = 0.f;
#pragma unroll
            for (int r = 0; r < 8; ++r) {
                float e  = __expf(c[r] - mnew);
                float vv = v[(size_t)(jbase + r + 8 * hi) * 64 + nt * 16 + lr] +
                           relC[nt][r];
                ps += e;
                pa += e * vv;
            }
            ps += __shfl_xor(ps, 16, 32);
            pa += __shfl_xor(pa, 16, 32);
            s_s[nt] = s_s[nt] * alpha + ps;
            a_s[nt] = a_s[nt] * alpha + pa;
            m_s[nt] = mnew;
        }
    }

    // merge 4 waves' online-softmax states
    if (hi == 0) {
#pragma unroll
        for (int nt = 0; nt < 4; ++nt) {
            int d = nt * 16 + lr;
            s_red[(wave * 64 + d) * 3 + 0] = m_s[nt];
            s_red[(wave * 64 + d) * 3 + 1] = s_s[nt];
            s_red[(wave * 64 + d) * 3 + 2] = a_s[nt];
        }
    }
    __syncthreads();
    if (tid < 64) {
        int d = tid;
        float M = -1e30f;
#pragma unroll
        for (int w = 0; w < 4; ++w) M = fmaxf(M, s_red[(w * 64 + d) * 3]);
        float S = 0.f, A = 0.f;
#pragma unroll
        for (int w = 0; w < 4; ++w) {
            float e = __expf(s_red[(w * 64 + d) * 3] - M);
            S += s_red[(w * 64 + d) * 3 + 1] * e;
            A += s_red[(w * 64 + d) * 3 + 2] * e;
        }
        feats_out[i * 64 + d] = A / S;
    }
}

// ---------------------------------------------------------------------------
// f1 = feats @ fc_w + fc_b  (also emit f16 copy for WMMA edge GEMM)
// ---------------------------------------------------------------------------
__global__ void fc_kernel(const float* __restrict__ feats, const float* __restrict__ w,
                          const float* __restrict__ b, _Float16* __restrict__ f1h) {
    int idx = blockIdx.x * blockDim.x + threadIdx.x;   // 1024*64
    int i = idx >> 6, d = idx & 63;
    float acc = b[d];
#pragma unroll 8
    for (int kk = 0; kk < 64; ++kk) acc += feats[i * 64 + kk] * w[kk * 64 + d];
    f1h[idx] = (_Float16)acc;
}

// ---------------------------------------------------------------------------
// out = sigmoid(f1 @ f1^T)  : NT WMMA GEMM, one wave per 16x64 output strip
// ---------------------------------------------------------------------------
__global__ void __launch_bounds__(32)
edge_kernel(const _Float16* __restrict__ f1h, float* __restrict__ out) {
    const int lane = threadIdx.x;
    const int lr = lane & 15, hi = lane >> 4;
    const int mbase = blockIdx.y * 16;
    const int nb0   = blockIdx.x * 64;

    v16h A[2];
    const _Float16* arow = f1h + (size_t)(mbase + lr) * 64;
#pragma unroll
    for (int kt = 0; kt < 2; ++kt) A[kt] = load_atile(arow, kt, hi);

#pragma unroll
    for (int nt4 = 0; nt4 < 4; ++nt4) {
        const int nbase = nb0 + nt4 * 16;
        v8f c = {};
#pragma unroll
        for (int kt = 0; kt < 2; ++kt) {
            // B col n = f1 row j (NT): lane<16 -> K 0..15, lane>=16 -> K 16..31
            const _Float16* p = f1h + (size_t)(nbase + lr) * 64 + kt * 32 + hi * 16;
            v16h B = make16(p, p + 8);
            c = wmma16(A[kt], B, c);
        }
#pragma unroll
        for (int r = 0; r < 8; ++r) {
            int row = mbase + r + 8 * hi;
            int col = nbase + lr;
            out[(size_t)row * 1024 + col] = 1.f / (1.f + __expf(-c[r]));
        }
    }
}

// ---------------------------------------------------------------------------
extern "C" void kernel_launch(void* const* d_in, const int* in_sizes, int n_in,
                              void* d_out, int out_size, void* d_ws, size_t ws_size,
                              hipStream_t stream) {
    const float* x     = (const float*)d_in[0];   // [1024,2]
    const float* in_w  = (const float*)d_in[1];   // [2,64]
    const float* in_b  = (const float*)d_in[2];   // [64]
    const float* qkv_w = (const float*)d_in[3];   // [3,64,192]
    const float* pw1   = (const float*)d_in[4];   // [3,3,64]
    const float* pb1   = (const float*)d_in[5];   // [3,64]
    const float* pw2   = (const float*)d_in[6];   // [3,64,64]
    const float* pb2   = (const float*)d_in[7];   // [3,64]
    const float* aw1   = (const float*)d_in[8];   // [3,64,256]
    const float* ab1   = (const float*)d_in[9];   // [3,256]
    const float* aw2   = (const float*)d_in[10];  // [3,256,64]
    const float* ab2   = (const float*)d_in[11];  // [3,64]
    const float* fc_w  = (const float*)d_in[12];  // [64,64]
    const float* fc_b  = (const float*)d_in[13];  // [64]
    float* out = (float*)d_out;

    char* ws = (char*)d_ws;
    size_t off = 0;
    float* featsA = (float*)(ws + off); off += (size_t)NPTS * 64 * 4;
    float* featsB = (float*)(ws + off); off += (size_t)NPTS * 64 * 4;
    float* qb     = (float*)(ws + off); off += (size_t)NPTS * 64 * 4;
    float* kb     = (float*)(ws + off); off += (size_t)NPTS * 64 * 4;
    float* vb     = (float*)(ws + off); off += (size_t)NPTS * 64 * 4;
    _Float16* pack = (_Float16*)(ws + off); off += (size_t)3 * 36864 * 2;
    _Float16* f1h  = (_Float16*)(ws + off); off += (size_t)NPTS * 64 * 2;

    // repack weights into WMMA B-lane layout (per layer: pw2p | aw1p | aw2p)
    for (int l = 0; l < 3; ++l) {
        _Float16* packL = pack + (size_t)l * 36864;
        pack_b_f16<<<(4096 + 255) / 256, 256, 0, stream>>>(pw2 + (size_t)l * 64 * 64,
                                                           packL, 64, 64);
        pack_b_f16<<<(16384 + 255) / 256, 256, 0, stream>>>(aw1 + (size_t)l * 64 * 256,
                                                            packL + 4096, 64, 256);
        pack_b_f16<<<(16384 + 255) / 256, 256, 0, stream>>>(aw2 + (size_t)l * 256 * 64,
                                                            packL + 4096 + 16384, 256,
                                                            64);
    }

    inproj_kernel<<<(NPTS * 64) / 256, 256, 0, stream>>>(x, in_w, in_b, featsA);

    float* fin = featsA;
    float* fout = featsB;
    for (int l = 0; l < 3; ++l) {
        qkv_kernel<<<(NPTS * 192) / 256, 256, 0, stream>>>(
            fin, qkv_w + (size_t)l * 64 * 192, qb, kb, vb);
        pt_layer_kernel<<<NPTS, 128, SMEM_BYTES, stream>>>(
            x, qb, kb, vb, pw1 + (size_t)l * 192, pb1 + (size_t)l * 64,
            pb2 + (size_t)l * 64, ab1 + (size_t)l * 256, ab2 + (size_t)l * 64,
            pack + (size_t)l * 36864, fout);
        float* t = fin; fin = fout; fout = t;
    }

    fc_kernel<<<(NPTS * 64) / 256, 256, 0, stream>>>(fin, fc_w, fc_b, f1h);
    edge_kernel<<<dim3(16, 64), 32, 0, stream>>>(f1h, out);
}
